// MultiHeadedAttentionWrapper_70342974374215
// MI455X (gfx1250) — matrix operand order repaired and checked by reading, not compiled, not run on hardware
//
#include <hip/hip_runtime.h>

// ---------------------------------------------------------------------------
// Multi-head causal attention for MI455X (gfx1250, wave32, WMMA).
// B=2, S=2048, D_IN=1024, H=16, HD=64.
//
// Kernel 1: per-head QKV projection GEMM, f32 -> f16 on the fly, 64-deep K
//           slabs through LDS, 8 back-to-back v_wmma_f32_16x16x32_f16 per
//           barrier pair, f16 Q/K/V to workspace [sel][h][b*S+s][HD].
// Kernel 2: flash attention (online softmax, fp32 stats/accum), QK^T and P*V
//           on WMMA, causal mask, 32-key chunks staged through LDS, all
//           matrix operands hoisted into registers before WMMA issue.
// ---------------------------------------------------------------------------

typedef _Float16 half_t;
typedef __attribute__((ext_vector_type(16))) _Float16 v16h;
typedef __attribute__((ext_vector_type(8)))  _Float16 v8h;
typedef __attribute__((ext_vector_type(4)))  _Float16 v4h;
typedef __attribute__((ext_vector_type(8)))  float    v8f;

#define D_IN  1024
#define HD    64
#define NH    16
#define SEQ   2048
#define NB    2
#define MROWS (NB * SEQ)   // 4096 rows in flattened [B,S]

static __device__ __forceinline__ v16h cat8(v8h a, v8h b) {
  return __builtin_shufflevector(a, b, 0, 1, 2, 3, 4, 5, 6, 7,
                                       8, 9, 10, 11, 12, 13, 14, 15);
}

static __device__ __forceinline__ v8f wmma_f16(v16h a, v16h b, v8f c) {
  // D = A(16x32 f16) * B(32x16 f16) + C(16x16 f32)
  return __builtin_amdgcn_wmma_f32_16x16x32_f16(
      /*neg_a=*/false, a, /*neg_b=*/false, b,
      /*c_mod=*/(short)0, c, /*reuse_a=*/false, /*reuse_b=*/false);
}

static __device__ __forceinline__ float rowmax16(float v) {
#pragma unroll
  for (int m = 1; m < 16; m <<= 1) v = fmaxf(v, __shfl_xor(v, m, 32));
  return v;
}

static __device__ __forceinline__ float rowsum16(float v) {
#pragma unroll
  for (int m = 1; m < 16; m <<= 1) v += __shfl_xor(v, m, 32);
  return v;
}

// ---------------------------------------------------------------------------
// Kernel 1: QKV projection.
// grid = (MROWS/128, NH, 3), block = 256 (8 waves).
// Each WG computes a 128(M) x 64(N) tile for one head; K staged in 64-slabs.
// ---------------------------------------------------------------------------
__global__ __launch_bounds__(256)
void qkv_gemm_kernel(const float* __restrict__ x,
                     const float* __restrict__ Wq,
                     const float* __restrict__ Wk,
                     const float* __restrict__ Wv,
                     half_t* __restrict__ qkv)
{
  __shared__ __align__(64) half_t Xs[128][64];   // [m][k]    16 KB
  __shared__ __align__(64) half_t WsT[64][64];   // [n][k]     8 KB (transposed)

  const int mtile = blockIdx.x;
  const int h     = blockIdx.y;
  const int sel   = blockIdx.z;
  const float* W  = (sel == 0) ? Wq : (sel == 1) ? Wk : Wv;
  const float* Wh = W + (size_t)h * D_IN * HD;

  const int tid  = threadIdx.x;
  const int wave = tid >> 5;
  const int lane = tid & 31;
  const int hi   = lane >> 4;   // lane half selects K sub-range (k = 16*hi + e, contiguous)
  const int ln   = lane & 15;

  const int xrow  = tid >> 1;   // 0..127 : X staging row
  const int xpart = tid & 1;    // which 32-half slab of the 64-wide K chunk
  const int wkrow = tid >> 2;   // 0..63  : W staging k-row
  const int wseg  = tid & 3;    // 16 output columns each

  v8f acc[4];
#pragma unroll
  for (int n = 0; n < 4; ++n)
#pragma unroll
    for (int i = 0; i < 8; ++i) acc[n][i] = 0.0f;

  for (int kk = 0; kk < D_IN; kk += 64) {
    // ---- stage X slab: f32 -> f16, row-major ----
    const float4* xs4 =
        (const float4*)(x + (size_t)(mtile * 128 + xrow) * D_IN + kk + xpart * 32);
#pragma unroll
    for (int i4 = 0; i4 < 8; ++i4) {
      float4 v = xs4[i4];
      v4h hv = {(half_t)v.x, (half_t)v.y, (half_t)v.z, (half_t)v.w};
      *(v4h*)&Xs[xrow][xpart * 32 + i4 * 4] = hv;
    }
    // ---- stage W slab transposed: WsT[n][k] = W[kk+k][n] ----
    const float4* ws4 = (const float4*)(Wh + (size_t)(kk + wkrow) * HD + wseg * 16);
#pragma unroll
    for (int i4 = 0; i4 < 4; ++i4) {
      float4 v = ws4[i4];
      const int c0 = wseg * 16 + i4 * 4;
      WsT[c0 + 0][wkrow] = (half_t)v.x;
      WsT[c0 + 1][wkrow] = (half_t)v.y;
      WsT[c0 + 2][wkrow] = (half_t)v.z;
      WsT[c0 + 3][wkrow] = (half_t)v.w;
    }
    __syncthreads();

    // ---- hoist all operands, then issue 8 WMMAs back-to-back ----
    // A (16x32 f16): lane ln = row; k = e + 16*hi
    const int arow = wave * 16 + ln;
    v16h A[2];
#pragma unroll
    for (int c = 0; c < 2; ++c) {
      v8h g0 = *(const v8h*)&Xs[arow][c * 32 + 16 * hi];
      v8h g1 = *(const v8h*)&Xs[arow][c * 32 + 16 * hi + 8];
      A[c] = cat8(g0, g1);
    }
    // B (32x16 f16): lane ln = column; k = e + 16*hi
    v16h Bn[4][2];
#pragma unroll
    for (int n = 0; n < 4; ++n)
#pragma unroll
      for (int c = 0; c < 2; ++c)
        Bn[n][c] = *(const v16h*)&WsT[n * 16 + ln][c * 32 + 16 * hi];

    // adjacent WMMAs hit different accumulators -> no back-to-back hazards
#pragma unroll
    for (int c = 0; c < 2; ++c)
#pragma unroll
      for (int n = 0; n < 4; ++n)
        acc[n] = wmma_f16(A[c], Bn[n][c], acc[n]);
    __syncthreads();
  }

  // ---- epilogue: f16 Q/K/V at [sel][h][row][hd] ----
  half_t* dst = qkv + (size_t)(sel * NH + h) * MROWS * HD;
#pragma unroll
  for (int n = 0; n < 4; ++n)
#pragma unroll
    for (int r = 0; r < 8; ++r) {
      const int srow = mtile * 128 + wave * 16 + r + 8 * hi;  // C layout rows
      dst[(size_t)srow * HD + n * 16 + ln] = (half_t)acc[n][r];
    }
}

// ---------------------------------------------------------------------------
// Kernel 2: causal flash attention over f16 Q/K/V, fp32 accumulation.
// grid = (SEQ/64, NH, NB), block = 128 (4 waves).
// Wave w handles query rows [q0 + 16w, q0 + 16w + 15]; key loop in 32-chunks,
// trip count uniform across the WG (fully-masked chunks contribute zeros).
// ---------------------------------------------------------------------------
__global__ __launch_bounds__(128)
void flash_attn_kernel(const half_t* __restrict__ qkv, float* __restrict__ out)
{
  __shared__ __align__(64) half_t Kt[32][64];      // [key][hd]       4 KB
  __shared__ __align__(64) half_t VtT[64][32];     // [hd][key]       4 KB
  __shared__ __align__(64) half_t Pb[4][16][32];   // per-wave P tile 4 KB

  const int qt = blockIdx.x;
  const int h  = blockIdx.y;
  const int b  = blockIdx.z;

  const int tid  = threadIdx.x;
  const int wave = tid >> 5;
  const int lane = tid & 31;
  const int hi   = lane >> 4;
  const int ln   = lane & 15;

  const half_t* Qp = qkv + ((size_t)(0 * NH + h) * MROWS + (size_t)b * SEQ) * HD;
  const half_t* Kp = qkv + ((size_t)(1 * NH + h) * MROWS + (size_t)b * SEQ) * HD;
  const half_t* Vp = qkv + ((size_t)(2 * NH + h) * MROWS + (size_t)b * SEQ) * HD;

  const int q0    = qt * 64;
  const int qrow0 = q0 + wave * 16;
  const float scale = 0.125f;  // 1/sqrt(HD)

  // ---- Q A-operands (two 32-deep hd chunks), straight from global ----
  const half_t* qr = Qp + (size_t)(qrow0 + ln) * HD;
  v16h Aq[2];
#pragma unroll
  for (int c = 0; c < 2; ++c) {
    v8h g0 = *(const v8h*)(qr + c * 32 + 16 * hi);
    v8h g1 = *(const v8h*)(qr + c * 32 + 16 * hi + 8);
    Aq[c] = cat8(g0, g1);
  }

  v8f O[4];
  float mR[8], lR[8];
#pragma unroll
  for (int n = 0; n < 4; ++n)
#pragma unroll
    for (int i = 0; i < 8; ++i) O[n][i] = 0.0f;
#pragma unroll
  for (int r = 0; r < 8; ++r) { mR[r] = -1e30f; lR[r] = 0.0f; }

  const int srow_k = tid >> 2;  // 0..31  staging key row
  const int seg    = tid & 3;   // 16-half segment of the 64-wide hd row

  const int nchunks = (q0 + 64) >> 5;  // uniform across the WG
  for (int j = 0; j < nchunks; ++j) {
    const int kb = j * 32;
    __syncthreads();  // previous iteration's LDS reads done

    // ---- stage K row-major, V transposed ----
    *(v16h*)&Kt[srow_k][seg * 16] =
        *(const v16h*)(Kp + (size_t)(kb + srow_k) * HD + seg * 16);
    v16h vv = *(const v16h*)(Vp + (size_t)(kb + srow_k) * HD + seg * 16);
#pragma unroll
    for (int i = 0; i < 16; ++i) VtT[seg * 16 + i][srow_k] = vv[i];

    if (j + 1 < nchunks) {  // pull next chunk toward the caches
      __builtin_prefetch(Kp + (size_t)(kb + 32 + srow_k) * HD + seg * 16, 0, 1);
      __builtin_prefetch(Vp + (size_t)(kb + 32 + srow_k) * HD + seg * 16, 0, 1);
    }
    __syncthreads();

    // ---- hoist ALL matrix operands for this chunk into registers ----
    v16h Bk[2][2];  // [key-half][hd-chunk]
#pragma unroll
    for (int c2 = 0; c2 < 2; ++c2) {
      Bk[c2][0] = *(const v16h*)&Kt[c2 * 16 + ln][16 * hi];        // hd 0..31
      Bk[c2][1] = *(const v16h*)&Kt[c2 * 16 + ln][32 + 16 * hi];   // hd 32..63
    }
    v16h Bv[4];  // P*V operands: loads overlap the softmax VALU below
#pragma unroll
    for (int n = 0; n < 4; ++n)
      Bv[n] = *(const v16h*)&VtT[n * 16 + ln][16 * hi];

    // ---- scores: S = Q K^T, 4 WMMAs, adjacent ones independent ----
    v8f s[2];
    {
      v8f z0, z1;
#pragma unroll
      for (int i = 0; i < 8; ++i) { z0[i] = 0.0f; z1[i] = 0.0f; }
      z0   = wmma_f16(Aq[0], Bk[0][0], z0);
      z1   = wmma_f16(Aq[0], Bk[1][0], z1);
      s[0] = wmma_f16(Aq[1], Bk[0][1], z0);
      s[1] = wmma_f16(Aq[1], Bk[1][1], z1);
    }

    // ---- causal mask + online softmax (C layout: row = r + 8*hi, col = ln) ----
#pragma unroll
    for (int r = 0; r < 8; ++r) {
      const int qidx = qrow0 + r + 8 * hi;
      float s0 = s[0][r] * scale;
      float s1 = s[1][r] * scale;
      s0 = ((kb + ln) <= qidx)      ? s0 : -1e30f;
      s1 = ((kb + 16 + ln) <= qidx) ? s1 : -1e30f;
      const float pm    = rowmax16(fmaxf(s0, s1));
      const float mn    = fmaxf(mR[r], pm);
      const float alpha = __expf(mR[r] - mn);
      const float p0    = __expf(s0 - mn);
      const float p1    = __expf(s1 - mn);
      const float rs    = rowsum16(p0 + p1);
      lR[r] = lR[r] * alpha + rs;
      mR[r] = mn;
#pragma unroll
      for (int n = 0; n < 4; ++n) O[n][r] *= alpha;
      // C layout -> LDS so it can be re-read in A layout
      Pb[wave][r + 8 * hi][ln]      = (half_t)p0;
      Pb[wave][r + 8 * hi][16 + ln] = (half_t)p1;
    }

    // same-wave DS ordering guarantee + compiler barrier
    asm volatile("s_wait_dscnt 0" ::: "memory");

    // ---- P as A operand (16x32): k = e + 16*hi ----
    v8h p0 = *(const v8h*)&Pb[wave][ln][16 * hi];
    v8h p1 = *(const v8h*)&Pb[wave][ln][16 * hi + 8];
    v16h Ap = cat8(p0, p1);

    // ---- O += P * V : B operands already in registers ----
#pragma unroll
    for (int n = 0; n < 4; ++n) O[n] = wmma_f16(Ap, Bv[n], O[n]);
  }

  // ---- epilogue: out[b][s][h*64 + col] = O / l ----
#pragma unroll
  for (int r = 0; r < 8; ++r) {
    const float inv  = 1.0f / lR[r];
    const int   srow = qrow0 + r + 8 * hi;
    float* orow = out + ((size_t)b * SEQ + srow) * (NH * HD) + h * HD;
#pragma unroll
    for (int n = 0; n < 4; ++n) orow[n * 16 + ln] = O[n][r] * inv;
  }
}

// ---------------------------------------------------------------------------
extern "C" void kernel_launch(void* const* d_in, const int* in_sizes, int n_in,
                              void* d_out, int out_size, void* d_ws, size_t ws_size,
                              hipStream_t stream) {
  (void)in_sizes; (void)n_in; (void)out_size;
  const float* x  = (const float*)d_in[0];
  const float* Wq = (const float*)d_in[1];
  const float* Wk = (const float*)d_in[2];
  const float* Wv = (const float*)d_in[3];
  float*  out = (float*)d_out;
  half_t* qkv = (half_t*)d_ws;  // [3][NH][MROWS][HD] f16, ~25 MB

  const size_t need = (size_t)3 * NH * MROWS * HD * sizeof(half_t);
  if (ws_size < need) return;

  dim3 g1(MROWS / 128, NH, 3);
  qkv_gemm_kernel<<<g1, 256, 0, stream>>>(x, Wq, Wk, Wv, qkv);

  dim3 g2(SEQ / 64, NH, NB);
  flash_attn_kernel<<<g2, 128, 0, stream>>>(qkv, out);
}
